// Net_EXTRA_22557168239431
// MI455X (gfx1250) — matrix operand order repaired and checked by reading, not compiled, not run on hardware
//
#include <hip/hip_runtime.h>
#include <hip/hip_bf16.h>

// ---------------------------------------------------------------------------
// Problem constants (from reference)
// ---------------------------------------------------------------------------
#define NN      100000
#define DD      32
#define EE      3200000
#define LAYERS  10
#define STEPSZ  0.001f
#define NEGSL   0.2f

typedef __attribute__((ext_vector_type(16))) _Float16 v16h;
typedef __attribute__((ext_vector_type(8)))  float    v8f;
typedef __attribute__((ext_vector_type(4)))  unsigned u32x4;
typedef __attribute__((ext_vector_type(8)))  int      i32x8;
typedef __attribute__((ext_vector_type(4)))  int      i32x4;

// order-preserving float->uint encoding for atomicMax-based segment max
__device__ __forceinline__ unsigned enc_f32(float f) {
    unsigned u = __float_as_uint(f);
    return (u & 0x80000000u) ? ~u : (u | 0x80000000u);
}
__device__ __forceinline__ float dec_f32(unsigned u) {
    return __uint_as_float((u & 0x80000000u) ? (u & 0x7FFFFFFFu) : ~u);
}
#define ENC_NEG_INF 0x007FFFFFu   // enc(-inf)

// ---------------------------------------------------------------------------
// Prep: swizzle A (f32 row-major 32x32 per node) into the exact CDNA5 WMMA
// f16 A-fragment layout (16x32 tiles, 2 tiles per node), so the matvec kernel
// reads one contiguous v16h (32B) per lane per tile.
//   lane<16 : M=lane,    K = i<8 ? i   : i+8
//   lane>=16: M=lane-16, K = i<8 ? i+8 : i+16
// ---------------------------------------------------------------------------
__global__ __launch_bounds__(256)
void swizzle_A_f16(const float* __restrict__ A, _Float16* __restrict__ Ah, int nNodes) {
    int wave = (int)((blockIdx.x * blockDim.x + threadIdx.x) >> 5);
    int lane = threadIdx.x & 31;
    if (wave >= nNodes) return;
    int mLocal = lane & 15;
    int kOff   = (lane >> 4) << 3;             // 0 or 8
    const float* Abase = A + (size_t)wave * (DD * DD);
    _Float16*    Dbase = Ah + (size_t)wave * (DD * DD);
#pragma unroll
    for (int t = 0; t < 2; ++t) {
        const float* row = Abase + (t * 16 + mLocal) * DD;
        _Float16*    dst = Dbase + t * 512 + lane * 16;
#pragma unroll
        for (int i = 0; i < 16; ++i) {
            int k = ((i < 8) ? i : i + 8) + kOff;
            dst[i] = (_Float16)row[k];
        }
    }
}

// ---------------------------------------------------------------------------
// Per-node 32x32 matvec via WMMA:  g[n] = 2 * A[n] @ (za[n] - sub*zb[n])
// One wave per node. The node's 2KB swizzled-f16 fragment block is staged
// Global->LDS by the Tensor Data Mover (tensor_load_to_lds, TENSORcnt), then
// two V_WMMA_F32_16X16X32_F16 (rows 0-15, 16-31) consume it from LDS.
// B = vector splatted into all 16 columns (lane holds row K=lane).
// ---------------------------------------------------------------------------
__global__ __launch_bounds__(256)
void quad_matvec_wmma(const _Float16* __restrict__ Ah,
                      const float* __restrict__ za,
                      const float* __restrict__ zb,
                      int sub,
                      float* __restrict__ g, int nNodes) {
    __shared__ _Float16 ldsA[8][DD * DD];   // 2KB per wave
    __shared__ float    ybuf[8][DD];        // result staging

    int wave = (int)((blockIdx.x * blockDim.x + threadIdx.x) >> 5);
    int w    = (threadIdx.x >> 5) & 7;
    int lane = threadIdx.x & 31;
    if (wave >= nNodes) return;   // uniform per wave: EXEC all-1s at WMMA

    // ---- TDM: DMA this node's 2KB fragment block into the wave's LDS slot ----
    unsigned long long ga = (unsigned long long)(uintptr_t)(Ah + (size_t)wave * (DD * DD));
    unsigned ldsa = (unsigned)(uintptr_t)(&ldsA[w][0]);   // flat addr[31:0] == LDS offset
    // D# group0: count=1 | lds_addr | global_addr[56:0] | type=2 ('10' in [127:126])
    u32x4 g0 = { 1u,
                 ldsa,
                 (unsigned)(ga & 0xFFFFFFFFu),
                 (unsigned)((ga >> 32) & 0x01FFFFFFu) | 0x80000000u };
    // D# group1: data_size=3 (8B) | tensor_dim0=256 | tensor_dim1=1 |
    //            tile_dim0=256 | tile_dim1=1 | tensor_dim0_stride=256
    i32x8 g1 = { (int)(3u << 16),
                 (int)(256u << 16),     // tensor_dim0[15:0] (=256) in bits [63:48]
                 (int)(1u << 16),       // tensor_dim0 hi=0, tensor_dim1=1
                 (int)(256u << 16),     // tensor_dim1 hi=0, tile_dim0=256
                 1,                     // tile_dim1=1, tile_dim2=0
                 256,                   // tensor_dim0_stride low32
                 0, 0 };
    i32x4 gz4 = { 0, 0, 0, 0 };
    i32x8 gz8 = { 0, 0, 0, 0, 0, 0, 0, 0 };
    __builtin_amdgcn_tensor_load_to_lds(g0, g1, gz4, gz4, gz8, 0);
    __builtin_amdgcn_s_wait_tensorcnt((short)0);
    asm volatile("" ::: "memory");

    // ---- build B: element i of lane holds w[K], K = (i<8 ? i : i+8) + 8*(lane>=16) ----
    float wv = za[(size_t)wave * DD + lane];
    if (sub) wv -= zb[(size_t)wave * DD + lane];
    v16h b;
#pragma unroll
    for (int i = 0; i < 16; ++i) b[i] = (_Float16)__shfl(wv, ((i < 8) ? i : i + 8) + ((lane >> 4) << 3), 32);

    const v16h* frag = (const v16h*)(&ldsA[w][0]);
    v16h a0 = frag[lane];        // tile 0: rows 0-15   (ds_load_b128 x2)
    v16h a1 = frag[32 + lane];   // tile 1: rows 16-31

    v8f c0 = {};
    v8f c1 = {};
    c0 = __builtin_amdgcn_wmma_f32_16x16x32_f16(false, a0, false, b, (short)0, c0, false, false);
    c1 = __builtin_amdgcn_wmma_f32_16x16x32_f16(false, a1, false, b, (short)0, c1, false, false);

    // D layout: VGPR r -> lanes 0-15: M=r, N=lane; lanes 16-31: M=r+8.
    // Column N==0 lives in lane 0 (M=0..7) and lane 16 (M=8..15).
    // Stage through LDS so the global store is a single coalesced b32 per lane.
    if (lane == 0) {
#pragma unroll
        for (int r = 0; r < 8; ++r) { ybuf[w][r]      = c0[r]; ybuf[w][16 + r] = c1[r]; }
    } else if (lane == 16) {
#pragma unroll
        for (int r = 0; r < 8; ++r) { ybuf[w][8 + r]  = c0[r]; ybuf[w][24 + r] = c1[r]; }
    }
    asm volatile("s_wait_dscnt 0x0" ::: "memory");   // same-wave LDS RAW
    g[(size_t)wave * DD + lane] = 2.0f * ybuf[w][lane];
}

// ---------------------------------------------------------------------------
// Attention pipeline
// ---------------------------------------------------------------------------
__global__ __launch_bounds__(256)
void node_dots(const float* __restrict__ x,
               const float* __restrict__ att_src,
               const float* __restrict__ att_dst,
               float* __restrict__ sArr, float* __restrict__ dArr, int nNodes) {
    int wave = (int)((blockIdx.x * blockDim.x + threadIdx.x) >> 5);
    int lane = threadIdx.x & 31;
    if (wave >= nNodes) return;
    float xv = x[(size_t)wave * DD + lane];
    float ps = xv * att_src[lane];
    float pd = xv * att_dst[lane];
#pragma unroll
    for (int off = 16; off > 0; off >>= 1) {
        ps += __shfl_down(ps, off, 32);
        pd += __shfl_down(pd, off, 32);
    }
    if (lane == 0) { sArr[wave] = ps; dArr[wave] = pd; }
}

__global__ __launch_bounds__(256)
void init_att(unsigned* __restrict__ mkey, float* __restrict__ den,
              unsigned* __restrict__ nnz, int nNodes) {
    int i = blockIdx.x * blockDim.x + threadIdx.x;
    if (i < nNodes) { mkey[i] = ENC_NEG_INF; den[i] = 0.0f; }
    if (i == 0) nnz[0] = 0u;
}

__global__ __launch_bounds__(256)
void edge_logit_max(const int* __restrict__ src, const int* __restrict__ dst,
                    const float* __restrict__ sArr, const float* __restrict__ dArr,
                    float* __restrict__ alpha, unsigned* __restrict__ mkey, int nEdges) {
    int e = blockIdx.x * blockDim.x + threadIdx.x;
    if (e >= nEdges) return;
    float val = sArr[src[e]] + dArr[dst[e]];
    val = (val > 0.0f) ? val : NEGSL * val;   // leaky_relu
    alpha[e] = val;
    atomicMax(&mkey[dst[e]], enc_f32(val));
}

__global__ __launch_bounds__(256)
void edge_exp_sum(const int* __restrict__ dst, const unsigned* __restrict__ mkey,
                  float* __restrict__ alpha, float* __restrict__ den, int nEdges) {
    int e = blockIdx.x * blockDim.x + threadIdx.x;
    if (e >= nEdges) return;
    int d = dst[e];
    float ex = __expf(alpha[e] - dec_f32(mkey[d]));
    alpha[e] = ex;
    atomicAdd(&den[d], ex);
}

__global__ __launch_bounds__(256)
void edge_norm_count(const int* __restrict__ dst, const float* __restrict__ den,
                     float* __restrict__ alpha, unsigned* __restrict__ nnz, int nEdges) {
    int e = blockIdx.x * blockDim.x + threadIdx.x;
    int lane = threadIdx.x & 31;
    float a = 0.0f;
    if (e < nEdges) {
        a = alpha[e] / (den[dst[e]] + 1e-16f);
        a = (a > 0.0f) ? a : 0.0f;            // threshold sparsification
        alpha[e] = a;
    }
    unsigned long long bal = __ballot(a > 0.0f);
    if (lane == 0 && bal) atomicAdd(nnz, (unsigned)__popcll(bal));
}

// ---------------------------------------------------------------------------
// Edge mixes: wave per edge, lane = feature (coalesced 128B rows).
// N*D state buffers fit in 192MB L2, so gathers/scatters are L2-resident.
// ---------------------------------------------------------------------------
__global__ __launch_bounds__(256)
void mix_one(const int* __restrict__ src, const int* __restrict__ dst,
             const float* __restrict__ alpha, const float* __restrict__ h,
             float* __restrict__ outmix, int nEdges) {
    int wave = (int)((blockIdx.x * blockDim.x + threadIdx.x) >> 5);
    int lane = threadIdx.x & 31;
    if (wave >= nEdges) return;
    float a = alpha[wave];
    if (a == 0.0f) return;
    int s = src[wave], d = dst[wave];
    atomicAdd(&outmix[(size_t)d * DD + lane], a * h[(size_t)s * DD + lane]);
}

__global__ __launch_bounds__(256)
void mix_three(const int* __restrict__ src, const int* __restrict__ dst,
               const float* __restrict__ alpha,
               const float* __restrict__ x1, const float* __restrict__ x0,
               const float* __restrict__ v,
               float* __restrict__ mix1, float* __restrict__ mix0,
               float* __restrict__ vnew, int nEdges) {
    int wave = (int)((blockIdx.x * blockDim.x + threadIdx.x) >> 5);
    int lane = threadIdx.x & 31;
    if (wave >= nEdges) return;
    float a = alpha[wave];
    if (a == 0.0f) return;
    int s = src[wave], d = dst[wave];
    atomicAdd(&mix1[(size_t)d * DD + lane], a * x1[(size_t)s * DD + lane]);
    atomicAdd(&mix0[(size_t)d * DD + lane], a * x0[(size_t)s * DD + lane]);
    if (lane == 0) atomicAdd(&vnew[d], a * v[s]);
}

// ---------------------------------------------------------------------------
// Elementwise kernels
// ---------------------------------------------------------------------------
__global__ __launch_bounds__(256)
void init_v_ones(float* __restrict__ v, int nNodes) {
    int i = blockIdx.x * blockDim.x + threadIdx.x;
    if (i < nNodes) v[i] = 1.0f;
}

__global__ __launch_bounds__(256)
void z_div(const float* __restrict__ x1, const float* __restrict__ v,
           float* __restrict__ z, int total) {
    int i = blockIdx.x * blockDim.x + threadIdx.x;
    if (i < total) z[i] = x1[i] / v[i >> 5];   // DD == 32
}

__global__ __launch_bounds__(256)
void update_layer0(const float* __restrict__ mix1, const float* __restrict__ g,
                   const float* __restrict__ b, float* __restrict__ xn, int total) {
    int i = blockIdx.x * blockDim.x + threadIdx.x;
    if (i < total) xn[i] = mix1[i] - STEPSZ * (g[i] + b[i]);
}

__global__ __launch_bounds__(256)
void update_layer(const float* __restrict__ mix1, const float* __restrict__ x1,
                  const float* __restrict__ mix0, const float* __restrict__ x0,
                  const float* __restrict__ g, float* __restrict__ xn, int total) {
    int i = blockIdx.x * blockDim.x + threadIdx.x;
    if (i < total)
        xn[i] = mix1[i] + x1[i] - 0.5f * (mix0[i] + x0[i]) - STEPSZ * g[i];
}

__global__ __launch_bounds__(256)
void finalize(const float* __restrict__ z, const unsigned* __restrict__ nnz,
              float* __restrict__ out, int total) {
    int i = blockIdx.x * blockDim.x + threadIdx.x;
    if (i < total) out[i] = z[i];
    if (i == 0) out[total] = (float)(nnz[0] * (unsigned)LAYERS);  // comm
}

// ---------------------------------------------------------------------------
// Host launcher.  Workspace layout (needs ~322 MB):
//   Ah (f16 swizzled A) : NN*1024*2           = 204.8 MB
//   alpha               : EE*4                =  12.8 MB
//   8 rotating N*D bufs : 8 * NN*DD*4         = 102.4 MB
//   s,d,den,mkey,v,vnew : 6 * NN*4            =   2.4 MB
//   nnz                 : 256 B
// ---------------------------------------------------------------------------
static inline char* carve(char*& p, size_t bytes) {
    char* r = p;
    p += (bytes + 255) & ~(size_t)255;
    return r;
}

extern "C" void kernel_launch(void* const* d_in, const int* in_sizes, int n_in,
                              void* d_out, int out_size, void* d_ws, size_t ws_size,
                              hipStream_t stream) {
    const float* x       = (const float*)d_in[0];
    const float* A       = (const float*)d_in[1];
    const float* b       = (const float*)d_in[2];
    const int*   ei      = (const int*)d_in[3];
    const float* att_src = (const float*)d_in[5];
    const float* att_dst = (const float*)d_in[6];
    const int* srcI = ei;
    const int* dstI = ei + EE;

    char* p = (char*)d_ws;
    _Float16* Ah    = (_Float16*)carve(p, (size_t)NN * DD * DD * sizeof(_Float16));
    float*    alpha = (float*)   carve(p, (size_t)EE * sizeof(float));
    float*    P[5];
    for (int i = 0; i < 5; ++i) P[i] = (float*)carve(p, (size_t)NN * DD * sizeof(float));
    float*    gbuf  = (float*)   carve(p, (size_t)NN * DD * sizeof(float));
    float*    mix1  = (float*)   carve(p, (size_t)NN * DD * sizeof(float));
    float*    mix0  = (float*)   carve(p, (size_t)NN * DD * sizeof(float));
    float*    sArr  = (float*)   carve(p, (size_t)NN * sizeof(float));
    float*    dArr  = (float*)   carve(p, (size_t)NN * sizeof(float));
    float*    den   = (float*)   carve(p, (size_t)NN * sizeof(float));
    unsigned* mkey  = (unsigned*)carve(p, (size_t)NN * sizeof(unsigned));
    float*    v     = (float*)   carve(p, (size_t)NN * sizeof(float));
    float*    vnew  = (float*)   carve(p, (size_t)NN * sizeof(float));
    unsigned* nnz   = (unsigned*)carve(p, 256);
    (void)ws_size; (void)in_sizes; (void)n_in; (void)out_size;

    const int TB = 256;
    const int nodeWaveBlocks = (NN + 7) / 8;        // wave-per-node kernels
    const int edgeWaveBlocks = (EE + 7) / 8;        // wave-per-edge kernels
    const int nodeThrBlocks  = (NN + TB - 1) / TB;
    const int edgeThrBlocks  = (EE + TB - 1) / TB;
    const int ndThrBlocks    = (NN * DD + TB - 1) / TB;
    const size_t ndBytes     = (size_t)NN * DD * sizeof(float);

    // ---- one-time prep (per call; deterministic) ----
    swizzle_A_f16<<<nodeWaveBlocks, TB, 0, stream>>>(A, Ah, NN);

    float* x1 = P[0]; float* x0 = P[1]; float* xn = P[2];
    float* z0 = P[3]; float* zB = P[4];
    (void)hipMemcpyAsync(x1, x, ndBytes, hipMemcpyDeviceToDevice, stream);
    (void)hipMemcpyAsync(z0, x, ndBytes, hipMemcpyDeviceToDevice, stream);
    init_v_ones<<<nodeThrBlocks, TB, 0, stream>>>(v, NN);

    // ---- attention (computed once, cached in alpha) ----
    node_dots<<<nodeWaveBlocks, TB, 0, stream>>>(x, att_src, att_dst, sArr, dArr, NN);
    init_att<<<nodeThrBlocks, TB, 0, stream>>>(mkey, den, nnz, NN);
    edge_logit_max<<<edgeThrBlocks, TB, 0, stream>>>(srcI, dstI, sArr, dArr, alpha, mkey, EE);
    edge_exp_sum<<<edgeThrBlocks, TB, 0, stream>>>(dstI, mkey, alpha, den, EE);
    edge_norm_count<<<edgeThrBlocks, TB, 0, stream>>>(dstI, den, alpha, nnz, EE);

    // ---- layer 0 :  xn = mix(x) - STEP*(2*A@x + b) ----
    (void)hipMemsetAsync(mix1, 0, ndBytes, stream);
    mix_one<<<edgeWaveBlocks, TB, 0, stream>>>(srcI, dstI, alpha, x1, mix1, EE);
    quad_matvec_wmma<<<nodeWaveBlocks, TB, 0, stream>>>(Ah, x1, x1, 0, gbuf, NN);
    update_layer0<<<ndThrBlocks, TB, 0, stream>>>(mix1, gbuf, b, xn, NN * DD);
    { float* t = x0; x0 = x1; x1 = xn; xn = t; }    // x0<-x, x1<-new

    // ---- layers 1..9 ----
    for (int layer = 1; layer < LAYERS; ++layer) {
        z_div<<<ndThrBlocks, TB, 0, stream>>>(x1, v, zB, NN * DD);
        // g1 - g0 = 2*A @ (z - z0)  -> single bandwidth pass over A per layer
        quad_matvec_wmma<<<nodeWaveBlocks, TB, 0, stream>>>(Ah, zB, z0, 1, gbuf, NN);
        (void)hipMemsetAsync(mix1, 0, ndBytes, stream);
        (void)hipMemsetAsync(mix0, 0, ndBytes, stream);
        (void)hipMemsetAsync(vnew, 0, (size_t)NN * sizeof(float), stream);
        mix_three<<<edgeWaveBlocks, TB, 0, stream>>>(srcI, dstI, alpha, x1, x0, v,
                                                     mix1, mix0, vnew, EE);
        update_layer<<<ndThrBlocks, TB, 0, stream>>>(mix1, x1, mix0, x0, gbuf, xn, NN * DD);
        { float* t = z0; z0 = zB; zB = t; }          // z0 <- current z
        { float* t = x0; x0 = x1; x1 = xn; xn = t; } // rotate x buffers
        { float* t = v;  v = vnew; vnew = t; }       // v <- mix(v)
    }

    // ---- output: z (last computed) + comm = LAYERS * nnz ----
    finalize<<<ndThrBlocks, TB, 0, stream>>>(z0, nnz, (float*)d_out, NN * DD);
}
